// SelfAttention2_55027120996797
// MI455X (gfx1250) — compile-verified, matrix operand order checked by reading, not run on hardware
//
#include <hip/hip_runtime.h>

// ---------------------------------------------------------------------------
// CDNA5 (gfx1250) fused self-attention: bf16 WMMA GEMMs + flash attention,
// with ASYNCcnt-tracked global->LDS tile staging (global_load_async_to_lds).
// B=4, L=2048, D=1024, H=16, DH=64
// ---------------------------------------------------------------------------

typedef __attribute__((ext_vector_type(16))) __bf16 v16bf;
typedef __attribute__((ext_vector_type(8)))  __bf16 v8bf;
typedef __attribute__((ext_vector_type(8)))  float  v8f;

union AFrag { v16bf v; v8bf h[2]; };

#define B_  4
#define L_  2048
#define D_  1024
#define H_  16
#define DH_ 64

static __device__ __forceinline__ v8f wmma_bf16(const AFrag& a, const AFrag& b, v8f c) {
  // D = A(16x32 bf16) * B(32x16 bf16) + C(16x16 f32)
  return __builtin_amdgcn_wmma_f32_16x16x32_bf16(false, a.v, false, b.v,
                                                 (short)0, c, false, false);
}

// Async global->LDS 16B copy (GV mode: 64-bit address in VGPR pair, no SADDR).
// VDST VGPR holds the per-lane LDS byte address; tracked by ASYNCcnt.
static __device__ __forceinline__ void async_ld_b128(const void* lds_dst, const void* gsrc) {
  const unsigned lds_off = (unsigned)(size_t)lds_dst;  // LDS aperture: addr[31:0] = LDS byte addr
  asm volatile("global_load_async_to_lds_b128 %0, %1, off"
               :: "v"(lds_off), "v"(gsrc) : "memory");
}

static __device__ __forceinline__ void wait_async0() {
  asm volatile("s_wait_asynccnt 0x0" ::: "memory");
}

// ---------------------------------------------------------------------------
// GEMM: Out = A(MxK) @ W(NxK)^T + bias   (bf16 WMMA, f32 accumulate)
// A_IS_BF16: A operand is bf16 (workspace, staged via async LDS DMA) vs f32.
// OUT_HEADSPLIT: write bf16 into [B,H,L,DH] layout; else write f32 row-major.
// Block: 256 threads (8 waves), tile 128x128, K-step 32.
// ---------------------------------------------------------------------------
template <bool A_IS_BF16, bool OUT_HEADSPLIT>
__global__ __launch_bounds__(256)
void gemm_bias_kernel(const void* __restrict__ Aptr, const float* __restrict__ W,
                      const float* __restrict__ bias, void* __restrict__ Outp,
                      int M, int N, int K) {
  __shared__ alignas(16) __bf16 lA[128 * 32];
  __shared__ alignas(16) __bf16 lB[128 * 32];

  const int tid  = threadIdx.x;
  const int lane = tid & 31;
  const int wave = tid >> 5;
  const int wr   = wave >> 1;    // 0..3 : wave row group (32 rows each)
  const int wc   = wave & 1;     // 0..1 : wave col group (64 cols each)
  const int rowBase = blockIdx.y * 128;
  const int colBase = blockIdx.x * 128;

  const v8f vz = {0.f, 0.f, 0.f, 0.f, 0.f, 0.f, 0.f, 0.f};
  v8f acc[2][4];
#pragma unroll
  for (int mt = 0; mt < 2; ++mt)
#pragma unroll
    for (int nt = 0; nt < 4; ++nt) acc[mt][nt] = vz;

  const int r  = tid >> 1;          // 0..127 staged row
  const int cb = (tid & 1) * 16;    // 0 or 16 within 32-wide K slice

  const float*  Af = (const float*)Aptr;
  const __bf16* Ab = (const __bf16*)Aptr;

  const int lr = lane & 15;
  const int kb = (lane < 16) ? 0 : 8;   // ISA 16-bit A/B fragment half-wave K base

  const int nK = K / 32;
  for (int kt = 0; kt < nK; ++kt) {
    const int k0 = kt * 32;

    // ---- stage A tile (128x32) as bf16 ----
    if (A_IS_BF16) {
      // Async DMA: global bf16 -> LDS, no VGPR round-trip (ASYNCcnt tracked).
      const __bf16* src = Ab + (size_t)(rowBase + r) * K + k0 + cb;
      async_ld_b128(&lA[r * 32 + cb],     src);
      async_ld_b128(&lA[r * 32 + cb + 8], src + 8);
      if (kt + 1 < nK) __builtin_prefetch(src + 32, 0, 1);
    } else {
      const float* src = Af + (size_t)(rowBase + r) * K + k0 + cb;
      float4 f0 = ((const float4*)src)[0];
      float4 f1 = ((const float4*)src)[1];
      float4 f2 = ((const float4*)src)[2];
      float4 f3 = ((const float4*)src)[3];
      __bf16* dst = &lA[r * 32 + cb];
      dst[0]  = (__bf16)f0.x; dst[1]  = (__bf16)f0.y; dst[2]  = (__bf16)f0.z; dst[3]  = (__bf16)f0.w;
      dst[4]  = (__bf16)f1.x; dst[5]  = (__bf16)f1.y; dst[6]  = (__bf16)f1.z; dst[7]  = (__bf16)f1.w;
      dst[8]  = (__bf16)f2.x; dst[9]  = (__bf16)f2.y; dst[10] = (__bf16)f2.z; dst[11] = (__bf16)f2.w;
      dst[12] = (__bf16)f3.x; dst[13] = (__bf16)f3.y; dst[14] = (__bf16)f3.z; dst[15] = (__bf16)f3.w;
      if (kt + 1 < nK) __builtin_prefetch(src + 32, 0, 1);
    }
    // ---- stage B tile (128x32) from W rows (W is [N,K] row-major) ----
    {
      const float* src = W + (size_t)(colBase + r) * K + k0 + cb;
      float4 f0 = ((const float4*)src)[0];
      float4 f1 = ((const float4*)src)[1];
      float4 f2 = ((const float4*)src)[2];
      float4 f3 = ((const float4*)src)[3];
      __bf16* dst = &lB[r * 32 + cb];
      dst[0]  = (__bf16)f0.x; dst[1]  = (__bf16)f0.y; dst[2]  = (__bf16)f0.z; dst[3]  = (__bf16)f0.w;
      dst[4]  = (__bf16)f1.x; dst[5]  = (__bf16)f1.y; dst[6]  = (__bf16)f1.z; dst[7]  = (__bf16)f1.w;
      dst[8]  = (__bf16)f2.x; dst[9]  = (__bf16)f2.y; dst[10] = (__bf16)f2.z; dst[11] = (__bf16)f2.w;
      dst[12] = (__bf16)f3.x; dst[13] = (__bf16)f3.y; dst[14] = (__bf16)f3.z; dst[15] = (__bf16)f3.w;
      if (kt + 1 < nK) __builtin_prefetch(src + 32, 0, 1);
    }
    if (A_IS_BF16) wait_async0();   // ASYNCcnt==0: async LDS writes landed
    __syncthreads();

    // ---- compute: 8 WMMAs per wave per K-step ----
    AFrag af[2], bf[4];
#pragma unroll
    for (int mt = 0; mt < 2; ++mt) {
      const int base = (wr * 32 + mt * 16 + lr) * 32;
      af[mt].h[0] = *(const v8bf*)&lA[base + kb];
      af[mt].h[1] = *(const v8bf*)&lA[base + kb + 16];
    }
#pragma unroll
    for (int nt = 0; nt < 4; ++nt) {
      const int base = (wc * 64 + nt * 16 + lr) * 32;
      bf[nt].h[0] = *(const v8bf*)&lB[base + kb];
      bf[nt].h[1] = *(const v8bf*)&lB[base + kb + 16];
    }
#pragma unroll
    for (int mt = 0; mt < 2; ++mt)
#pragma unroll
      for (int nt = 0; nt < 4; ++nt)
        acc[mt][nt] = wmma_bf16(af[mt], bf[nt], acc[mt][nt]);
    __syncthreads();
  }

  // ---- epilogue: C/D layout -> (row = j + 8*(lane>=16), col = lane&15) ----
  const int rsel = (lane >> 4) * 8;
#pragma unroll
  for (int mt = 0; mt < 2; ++mt)
#pragma unroll
    for (int nt = 0; nt < 4; ++nt)
#pragma unroll
      for (int j = 0; j < 8; ++j) {
        const int row = rowBase + wr * 32 + mt * 16 + rsel + j;
        const int col = colBase + wc * 64 + nt * 16 + lr;
        const float val = acc[mt][nt][j] + bias[col];
        if (OUT_HEADSPLIT) {
          const int bb = row / L_, ll = row % L_;
          const int hh = col / DH_, dd = col % DH_;
          ((__bf16*)Outp)[(((size_t)(bb * H_ + hh)) * L_ + ll) * DH_ + dd] = (__bf16)val;
        } else {
          ((float*)Outp)[(size_t)row * N + col] = val;
        }
      }
}

// ---------------------------------------------------------------------------
// Flash attention: one block handles 128 query rows of one (b,h).
// 8 waves, wave w owns query rows [w*16, w*16+16). K/V tiles of 64 keys in LDS.
// K tile staged via async LDS DMA; V staged transposed; online softmax in f32;
// S and PV products via bf16 WMMA.
// ---------------------------------------------------------------------------
__global__ __launch_bounds__(256)
void flash_attn_kernel(const __bf16* __restrict__ Q, const __bf16* __restrict__ Kb,
                       const __bf16* __restrict__ Vb, const unsigned char* __restrict__ mask,
                       __bf16* __restrict__ Out) {
  __shared__ alignas(16) __bf16 lK[64 * 64];    // [key][dh]
  __shared__ alignas(16) __bf16 lVt[64 * 64];   // [dh][key]  (transposed at stage)
  __shared__ alignas(16) __bf16 lP[128 * 64];   // per-wave 16x64 P tiles

  const int tid  = threadIdx.x;
  const int lane = tid & 31;
  const int w    = tid >> 5;
  const int bh   = blockIdx.y;
  const int b    = bh >> 4;          // H_ == 16
  const int h    = bh & 15;
  const int qbase = blockIdx.x * 128;

  const __bf16* Qh = Q  + (size_t)bh * L_ * DH_;
  const __bf16* Kh = Kb + (size_t)bh * L_ * DH_;
  const __bf16* Vh = Vb + (size_t)bh * L_ * DH_;
  const unsigned char* mb = mask + (size_t)b * L_;

  const int lr   = lane & 15;
  const int kb   = (lane < 16) ? 0 : 8;
  const int rsel = (lane >> 4) * 8;

  // Q A-fragments for the wave's 16 rows: 2 chunks over DH=64
  AFrag aQ[2];
#pragma unroll
  for (int c = 0; c < 2; ++c) {
    const __bf16* qp = Qh + (size_t)(qbase + w * 16 + lr) * DH_ + c * 32;
    aQ[c].h[0] = *(const v8bf*)(qp + kb);
    aQ[c].h[1] = *(const v8bf*)(qp + kb + 16);
  }

  const v8f vz = {0.f, 0.f, 0.f, 0.f, 0.f, 0.f, 0.f, 0.f};
  v8f O[4];
#pragma unroll
  for (int nt = 0; nt < 4; ++nt) O[nt] = vz;
  float mrow[8], lsum[8];
#pragma unroll
  for (int j = 0; j < 8; ++j) { mrow[j] = -1e30f; lsum[j] = 0.f; }

  const int kt_end = (qbase + 127) / 64;   // causal: only tiles with keys <= last query
  for (int kt = 0; kt <= kt_end; ++kt) {
    const int ks = kt * 64;
    // ---- stage K tile (async global->LDS DMA) and transposed V tile ----
    {
      const int r   = tid >> 2;          // 0..63 key row
      const int cb2 = (tid & 3) * 16;    // dh offset 0/16/32/48
      const __bf16* kp = Kh + (size_t)(ks + r) * DH_ + cb2;
      async_ld_b128(&lK[r * 64 + cb2],     kp);
      async_ld_b128(&lK[r * 64 + cb2 + 8], kp + 8);
      const __bf16* vp = Vh + (size_t)(ks + r) * DH_ + cb2;
#pragma unroll
      for (int i = 0; i < 16; ++i) lVt[(cb2 + i) * 64 + r] = vp[i];
      if (kt < kt_end) {
        __builtin_prefetch(kp + 64 * DH_, 0, 1);
        __builtin_prefetch(vp + 64 * DH_, 0, 1);
      }
    }
    wait_async0();                       // ASYNCcnt==0: K tile resident in LDS
    __syncthreads();

    // ---- S = Q * K^T  (16x64 per wave, 8 WMMAs) ----
    v8f s[4];
#pragma unroll
    for (int nt = 0; nt < 4; ++nt) s[nt] = vz;
#pragma unroll
    for (int nt = 0; nt < 4; ++nt)
#pragma unroll
      for (int c = 0; c < 2; ++c) {
        AFrag bk_;
        const int base = (nt * 16 + lr) * 64 + c * 32;
        bk_.h[0] = *(const v8bf*)&lK[base + kb];
        bk_.h[1] = *(const v8bf*)&lK[base + kb + 16];
        s[nt] = wmma_bf16(aQ[c], bk_, s[nt]);
      }

    // ---- scale + causal & padding mask ----
    const float scale = 0.125f;   // 1/sqrt(DH)
#pragma unroll
    for (int nt = 0; nt < 4; ++nt) {
      const int colg = ks + nt * 16 + lr;
      const bool kvalid = (mb[colg] != 0);
#pragma unroll
      for (int j = 0; j < 8; ++j) {
        const int rowg = qbase + w * 16 + rsel + j;
        const float sv = s[nt][j] * scale;
        s[nt][j] = (kvalid && colg <= rowg) ? sv : -1e30f;
      }
    }

    // ---- online softmax (row-wise, 16-lane butterfly reductions) ----
#pragma unroll
    for (int j = 0; j < 8; ++j) {
      float mx = fmaxf(fmaxf(s[0][j], s[1][j]), fmaxf(s[2][j], s[3][j]));
#pragma unroll
      for (int off = 8; off >= 1; off >>= 1) mx = fmaxf(mx, __shfl_xor(mx, off, 32));
      const float mnew  = fmaxf(mrow[j], mx);
      const float alpha = __expf(mrow[j] - mnew);
      mrow[j] = mnew;
      float psum = 0.f;
#pragma unroll
      for (int nt = 0; nt < 4; ++nt) {
        const float sv = s[nt][j];
        const float p  = (sv > -1e29f) ? __expf(sv - mnew) : 0.f;
        s[nt][j] = p;
        psum += p;
      }
#pragma unroll
      for (int off = 8; off >= 1; off >>= 1) psum += __shfl_xor(psum, off, 32);
      lsum[j] = lsum[j] * alpha + psum;
#pragma unroll
      for (int nt = 0; nt < 4; ++nt) O[nt][j] *= alpha;
    }

    // ---- P -> LDS (wave-private; LDS is in-order per wave) ----
#pragma unroll
    for (int nt = 0; nt < 4; ++nt)
#pragma unroll
      for (int j = 0; j < 8; ++j)
        lP[(w * 16 + rsel + j) * 64 + nt * 16 + lr] = (__bf16)s[nt][j];

    // ---- O += P(16x64) * V(64x64)  (8 WMMAs) ----
#pragma unroll
    for (int c = 0; c < 2; ++c) {
      AFrag aP;
      const int pbase = (w * 16 + lr) * 64 + c * 32;
      aP.h[0] = *(const v8bf*)&lP[pbase + kb];
      aP.h[1] = *(const v8bf*)&lP[pbase + kb + 16];
#pragma unroll
      for (int nt = 0; nt < 4; ++nt) {
        AFrag bv_;
        const int base = (nt * 16 + lr) * 64 + c * 32;
        bv_.h[0] = *(const v8bf*)&lVt[base + kb];
        bv_.h[1] = *(const v8bf*)&lVt[base + kb + 16];
        O[nt] = wmma_bf16(aP, bv_, O[nt]);
      }
    }
    __syncthreads();
  }

  // ---- normalize + write bf16 [B*L, D] with heads re-interleaved ----
#pragma unroll
  for (int nt = 0; nt < 4; ++nt)
#pragma unroll
    for (int j = 0; j < 8; ++j) {
      const int rowg = qbase + w * 16 + rsel + j;
      const int col  = h * DH_ + nt * 16 + lr;
      const float ov = O[nt][j] / fmaxf(lsum[j], 1e-20f);
      Out[(size_t)(b * L_ + rowg) * D_ + col] = (__bf16)ov;
    }
}

// ---------------------------------------------------------------------------
// Launch: QKV projections -> flash attention -> output projection.
// ---------------------------------------------------------------------------
extern "C" void kernel_launch(void* const* d_in, const int* in_sizes, int n_in,
                              void* d_out, int out_size, void* d_ws, size_t ws_size,
                              hipStream_t stream) {
  (void)in_sizes; (void)n_in; (void)out_size; (void)ws_size;

  const float* q  = (const float*)d_in[0];
  const float* k  = (const float*)d_in[1];
  const float* v  = (const float*)d_in[2];
  const unsigned char* am = (const unsigned char*)d_in[3];   // jax bool mask
  const float* Wq = (const float*)d_in[4];
  const float* bq = (const float*)d_in[5];
  const float* Wk = (const float*)d_in[6];
  const float* bk = (const float*)d_in[7];
  const float* Wv = (const float*)d_in[8];
  const float* bv = (const float*)d_in[9];
  const float* Wc = (const float*)d_in[10];
  const float* bc = (const float*)d_in[11];

  char* ws = (char*)d_ws;
  const size_t seg = (size_t)B_ * H_ * L_ * DH_ * sizeof(__bf16);   // 16 MB
  __bf16* Qbf = (__bf16*)(ws);
  __bf16* Kbf = (__bf16*)(ws + seg);
  __bf16* Vbf = (__bf16*)(ws + 2 * seg);
  __bf16* Abf = (__bf16*)(ws + 3 * seg);

  const int M = B_ * L_;             // 8192
  dim3 blk(256);
  dim3 gproj(D_ / 128, M / 128);     // (8, 64)
  gemm_bias_kernel<false, true><<<gproj, blk, 0, stream>>>(q, Wq, bq, Qbf, M, D_, D_);
  gemm_bias_kernel<false, true><<<gproj, blk, 0, stream>>>(k, Wk, bk, Kbf, M, D_, D_);
  gemm_bias_kernel<false, true><<<gproj, blk, 0, stream>>>(v, Wv, bv, Vbf, M, D_, D_);

  dim3 gattn(L_ / 128, B_ * H_);     // (16, 64)
  flash_attn_kernel<<<gattn, blk, 0, stream>>>(Qbf, Kbf, Vbf, am, Abf);

  gemm_bias_kernel<true, false><<<gproj, blk, 0, stream>>>(Abf, Wc, bc, (float*)d_out, M, D_, D_);
}
